// RelMHAttn_19129784336860
// MI455X (gfx1250) — matrix operand order repaired and checked by reading, not compiled, not run on hardware
//
#include <hip/hip_runtime.h>
#include <hip/hip_bf16.h>
#include <stdint.h>

// ---------------------------------------------------------------------------
// Relative-position MHA for MI455X (gfx1250): bf16 WMMA everywhere,
// flash-attention inner loop, 2x2 register blocking in GEMMs, V stored
// transposed so every WMMA operand tile loads as global_load_b128 pairs.
// ---------------------------------------------------------------------------

namespace {
constexpr int kB      = 4;
constexpr int kT      = 2048;
constexpr int kD      = 768;
constexpr int kH      = 12;
constexpr int kHD     = 64;
constexpr int kMaxRel = 128;
constexpr int kNumRel = 2 * kMaxRel + 1;  // 257
constexpr int kM      = kB * kT;          // 8192 rows for projections
constexpr int kMT32   = kM / 32;          // 256 row blocks (32 rows)
constexpr int kNT32   = kD / 32;          // 24 col blocks (32 cols)
}

typedef __attribute__((ext_vector_type(16))) __bf16 v16bf;
typedef __attribute__((ext_vector_type(8)))  __bf16 bf16x8;
typedef __attribute__((ext_vector_type(8)))  float  v8f;

// D = A(16x32) * B(32x16) + C, bf16 in, f32 accumulate
static __device__ __forceinline__ v8f wmma_bf16(v16bf a, v16bf b, v8f c) {
  return __builtin_amdgcn_wmma_f32_16x16x32_bf16(
      /*neg_a=*/false, a, /*neg_b=*/false, b,
      /*c_mod=*/(short)0, c, /*reuse_a=*/false, /*reuse_b=*/false);
}

// A-matrix 16x32 bf16 layout (ISA 7.12.2): lane L (row = L&15, half = L>>4)
// VGPR 0-3 hold K = half*8 + {0..7}; VGPR 4-7 hold K = 16 + half*8 + {0..7}.
static __device__ __forceinline__ v16bf load_a_cont(const __bf16* p, int stride_m, int lane) {
  const int row = lane & 15, half = lane >> 4;
  const __bf16* base = p + (size_t)row * stride_m + half * 8;
  bf16x8 lo = *reinterpret_cast<const bf16x8*>(base);       // K = half*8 .. +7
  bf16x8 hi = *reinterpret_cast<const bf16x8*>(base + 16);  // K = 16+half*8 .. +7
  v16bf r;
#pragma unroll
  for (int j = 0; j < 8; ++j) { r[j] = lo[j]; r[8 + j] = hi[j]; }
  return r;
}

// B-matrix 32x16 bf16 layout: lane L (col = L&15, half = L>>4);
// VGPR 0-3 hold K = half*8 + {0..7}; VGPR 4-7 hold K = 16 + half*8 + {0..7}; elem(n,k) = p[n*stride_n + k].
static __device__ __forceinline__ v16bf load_b_cont(const __bf16* p, size_t stride_n, int lane) {
  const int col = lane & 15, half = lane >> 4;
  const __bf16* base = p + (size_t)col * stride_n + half * 8;
  bf16x8 lo = *reinterpret_cast<const bf16x8*>(base);
  bf16x8 hi = *reinterpret_cast<const bf16x8*>(base + 16);
  v16bf r;
#pragma unroll
  for (int j = 0; j < 8; ++j) { r[j] = lo[j]; r[8 + j] = hi[j]; }
  return r;
}

// reductions across the 16 lanes of a half-wave (xor masks keep bit4 fixed,
// matching the C-tile layout where rows live in half-waves)
static __device__ __forceinline__ float halfmax16(float v) {
#pragma unroll
  for (int m = 1; m < 16; m <<= 1) v = fmaxf(v, __shfl_xor(v, m, 32));
  return v;
}
static __device__ __forceinline__ float halfsum16(float v) {
#pragma unroll
  for (int m = 1; m < 16; m <<= 1) v += __shfl_xor(v, m, 32);
  return v;
}

// ---------------------------------------------------------------------------
// 1) relative-bias head mean: mask257[d] = mean_h bias_table[d, h]
__global__ void bias_mean_kernel(const float* __restrict__ bias_table,
                                 float* __restrict__ mask257) {
  int i = blockIdx.x * blockDim.x + threadIdx.x;
  if (i < kNumRel) {
    float s = 0.f;
#pragma unroll
    for (int h = 0; h < kH; ++h) s += bias_table[i * kH + h];
    mask257[i] = s * (1.0f / kH);
  }
}

// 2) fp32 -> bf16 conversion
__global__ void cvt_f32_bf16_kernel(const float* __restrict__ in,
                                    __bf16* __restrict__ out, int n) {
  int i = blockIdx.x * blockDim.x + threadIdx.x;
  if (i < n) out[i] = (__bf16)in[i];
}

// ---------------------------------------------------------------------------
// 3) fused Q/K/V projection: y = x @ W^T + b; 2x2 tile blocking per wave.
//    Q,K scattered to [B,H,T,HD]; V scattered TRANSPOSED to [B,H,HD,T].
__global__ void proj_qkv_kernel(const __bf16* __restrict__ xb,
                                const __bf16* __restrict__ wqb,
                                const __bf16* __restrict__ wkb,
                                const __bf16* __restrict__ wvb,
                                const float* __restrict__ bq,
                                const float* __restrict__ bk,
                                const float* __restrict__ bv,
                                __bf16* __restrict__ qo,
                                __bf16* __restrict__ ko,
                                __bf16* __restrict__ vTo) {
  const int z = blockIdx.z;
  const __bf16* W  = (z == 0) ? wqb : (z == 1) ? wkb : wvb;
  const float*  bi = (z == 0) ? bq  : (z == 1) ? bk  : bv;

  const int warp = threadIdx.x >> 5;
  const int lane = threadIdx.x & 31;
  const int tile = blockIdx.x * 8 + warp;
  const int mt = tile / kNT32, nt = tile % kNT32;
  const int m0 = mt * 32, n0 = nt * 32;

  v8f acc[2][2] = {};
  for (int kk = 0; kk < kD; kk += 32) {
    v16bf a0 = load_a_cont(xb + (size_t)m0 * kD + kk,        kD, lane);
    v16bf a1 = load_a_cont(xb + (size_t)(m0 + 16) * kD + kk, kD, lane);
    v16bf b0 = load_b_cont(W  + (size_t)n0 * kD + kk,        kD, lane);
    v16bf b1 = load_b_cont(W  + (size_t)(n0 + 16) * kD + kk, kD, lane);
    acc[0][0] = wmma_bf16(a0, b0, acc[0][0]);
    acc[0][1] = wmma_bf16(a0, b1, acc[0][1]);
    acc[1][0] = wmma_bf16(a1, b0, acc[1][0]);
    acc[1][1] = wmma_bf16(a1, b1, acc[1][1]);
  }

  const int half = lane >> 4, ncol = lane & 15;
#pragma unroll
  for (int mi = 0; mi < 2; ++mi) {
#pragma unroll
    for (int ni = 0; ni < 2; ++ni) {
      const int ng = n0 + ni * 16 + ncol;
      const float bias = bi[ng];
      const int h = ng >> 6, hd = ng & 63;
#pragma unroll
      for (int j = 0; j < 8; ++j) {
        const int mg = m0 + mi * 16 + j + 8 * half;
        const int b_ = mg >> 11;            // / kT
        const int t_ = mg & (kT - 1);
        const __bf16 val = (__bf16)(acc[mi][ni][j] + bias);
        if (z == 2) {
          vTo[(((size_t)(b_ * kH + h)) * kHD + hd) * kT + t_] = val;   // [B,H,HD,T]
        } else {
          __bf16* dst = (z == 0) ? qo : ko;
          dst[(((size_t)(b_ * kH + h)) * kT + t_) * kHD + hd] = val;   // [B,H,T,HD]
        }
      }
    }
  }
}

// ---------------------------------------------------------------------------
// 4) flash attention: one wave = 32 query rows of one (b,h); online softmax
//    over 2048 keys in blocks of 32.  K/V tiles shared by both 16-row q tiles.
__global__ void attn_kernel(const __bf16* __restrict__ q,
                            const __bf16* __restrict__ k,
                            const __bf16* __restrict__ vT,
                            const float* __restrict__ mask257,
                            __bf16* __restrict__ ctx) {
  __shared__ float  smask[kNumRel];
  __shared__ __bf16 pbuf[8 * 2 * 16 * 32];  // per-wave, per-qtile P staging

  for (int i = threadIdx.x; i < kNumRel; i += blockDim.x) smask[i] = mask257[i];
  __syncthreads();

  const int warp = threadIdx.x >> 5;
  const int lane = threadIdx.x & 31;
  const int half = lane >> 4, ncol = lane & 15;

  const int gw = blockIdx.x * 8 + warp;
  const int QT = kT / 32;               // 64 query blocks per head
  const int qt = gw % QT;
  const int bh = gw / QT;
  const int q0 = qt * 32;

  const __bf16* Qh  = q  + (size_t)bh * kT * kHD;
  const __bf16* Kh  = k  + (size_t)bh * kT * kHD;
  const __bf16* VTh = vT + (size_t)bh * kHD * kT;

  // Q block (32 x 64) as 2x2 A tiles, resident for the whole key loop
  v16bf aq[2][2];
#pragma unroll
  for (int qi = 0; qi < 2; ++qi) {
    aq[qi][0] = load_a_cont(Qh + (size_t)(q0 + qi * 16) * kHD,      kHD, lane);
    aq[qi][1] = load_a_cont(Qh + (size_t)(q0 + qi * 16) * kHD + 32, kHD, lane);
  }

  v8f acc[2][4] = {};                   // per q tile: 16 x 64 f32 accumulator
  float rmax[2][8], rsum[2][8];
#pragma unroll
  for (int qi = 0; qi < 2; ++qi)
#pragma unroll
    for (int j = 0; j < 8; ++j) { rmax[qi][j] = -1e30f; rsum[qi][j] = 0.f; }

  __bf16* pw = pbuf + warp * (2 * 16 * 32);

  for (int kb = 0; kb < kT; kb += 32) {
    // ---- K tiles for this 32-key block (shared by both q tiles)
    v16bf bk00 = load_b_cont(Kh + (size_t)kb * kHD,             kHD, lane);
    v16bf bk01 = load_b_cont(Kh + (size_t)kb * kHD + 32,        kHD, lane);
    v16bf bk10 = load_b_cont(Kh + (size_t)(kb + 16) * kHD,      kHD, lane);
    v16bf bk11 = load_b_cont(Kh + (size_t)(kb + 16) * kHD + 32, kHD, lane);

    float alpha[2][8];
#pragma unroll
    for (int qi = 0; qi < 2; ++qi) {
      // ---- scores: two 16x16 tiles (keys kb..+15, kb+16..+31)
      v8f s0 = {}, s1 = {};
      s0 = wmma_bf16(aq[qi][0], bk00, s0);
      s0 = wmma_bf16(aq[qi][1], bk01, s0);
      s1 = wmma_bf16(aq[qi][0], bk10, s1);
      s1 = wmma_bf16(aq[qi][1], bk11, s1);

      // ---- scale + relative-position bias
#pragma unroll
      for (int j = 0; j < 8; ++j) {
        const int qrow = q0 + qi * 16 + j + 8 * half;
        int d0 = qrow - (kb + ncol);
        int d1 = qrow - (kb + 16 + ncol);
        d0 = (d0 < -kMaxRel) ? -kMaxRel : (d0 > kMaxRel) ? kMaxRel : d0;
        d1 = (d1 < -kMaxRel) ? -kMaxRel : (d1 > kMaxRel) ? kMaxRel : d1;
        s0[j] = s0[j] * 0.125f + smask[d0 + kMaxRel];
        s1[j] = s1[j] * 0.125f + smask[d1 + kMaxRel];
      }

      // ---- online softmax update (rows distributed as j + 8*half)
#pragma unroll
      for (int j = 0; j < 8; ++j) {
        float tmax = halfmax16(fmaxf(s0[j], s1[j]));
        float nm   = fmaxf(rmax[qi][j], tmax);
        alpha[qi][j] = __expf(rmax[qi][j] - nm);
        rmax[qi][j]  = nm;
        float p0 = __expf(s0[j] - nm);
        float p1 = __expf(s1[j] - nm);
        s0[j] = p0; s1[j] = p1;
        rsum[qi][j] = rsum[qi][j] * alpha[qi][j] + halfsum16(p0 + p1);
      }

      // ---- stage P (C-layout) into LDS for A-layout reload
      __bf16* pq = pw + qi * (16 * 32);
#pragma unroll
      for (int j = 0; j < 8; ++j) {
        const int m = j + 8 * half;
        pq[m * 32 + ncol]      = (__bf16)s0[j];
        pq[m * 32 + 16 + ncol] = (__bf16)s1[j];
      }
    }

    __syncthreads();
    v16bf ap0 = load_a_cont(pw,           32, lane);
    v16bf ap1 = load_a_cont(pw + 16 * 32, 32, lane);
    __syncthreads();

    // ---- O = O*alpha + P @ V   (V^T: elem(n=hd, k=key) contiguous in key)
#pragma unroll
    for (int tt = 0; tt < 4; ++tt) {
      v16bf bv = load_b_cont(VTh + (size_t)(tt * 16) * kT + kb, kT, lane);
#pragma unroll
      for (int j = 0; j < 8; ++j) {
        acc[0][tt][j] *= alpha[0][j];
        acc[1][tt][j] *= alpha[1][j];
      }
      acc[0][tt] = wmma_bf16(ap0, bv, acc[0][tt]);
      acc[1][tt] = wmma_bf16(ap1, bv, acc[1][tt]);
    }
  }

  // ---- normalize and scatter ctx as [B,T,D] bf16 (n = h*64 + tt*16 + ncol)
  const int b_ = bh / kH, h_ = bh % kH;
#pragma unroll
  for (int qi = 0; qi < 2; ++qi)
#pragma unroll
    for (int tt = 0; tt < 4; ++tt)
#pragma unroll
      for (int j = 0; j < 8; ++j) {
        const int m = qi * 16 + j + 8 * half;
        const float o = acc[qi][tt][j] / rsum[qi][j];
        ctx[((size_t)(b_ * kT + q0 + m)) * kD + h_ * kHD + tt * 16 + ncol] = (__bf16)o;
      }
}

// ---------------------------------------------------------------------------
// 5) output projection: out = ctx @ Wo^T + bo (fp32 result), 2x2 blocking.
__global__ void out_proj_kernel(const __bf16* __restrict__ ctxb,
                                const __bf16* __restrict__ wob,
                                const float* __restrict__ bo,
                                float* __restrict__ out) {
  const int warp = threadIdx.x >> 5;
  const int lane = threadIdx.x & 31;
  const int tile = blockIdx.x * 8 + warp;
  const int mt = tile / kNT32, nt = tile % kNT32;
  const int m0 = mt * 32, n0 = nt * 32;

  v8f acc[2][2] = {};
  for (int kk = 0; kk < kD; kk += 32) {
    v16bf a0 = load_a_cont(ctxb + (size_t)m0 * kD + kk,        kD, lane);
    v16bf a1 = load_a_cont(ctxb + (size_t)(m0 + 16) * kD + kk, kD, lane);
    v16bf b0 = load_b_cont(wob  + (size_t)n0 * kD + kk,        kD, lane);
    v16bf b1 = load_b_cont(wob  + (size_t)(n0 + 16) * kD + kk, kD, lane);
    acc[0][0] = wmma_bf16(a0, b0, acc[0][0]);
    acc[0][1] = wmma_bf16(a0, b1, acc[0][1]);
    acc[1][0] = wmma_bf16(a1, b0, acc[1][0]);
    acc[1][1] = wmma_bf16(a1, b1, acc[1][1]);
  }

  const int half = lane >> 4, ncol = lane & 15;
#pragma unroll
  for (int mi = 0; mi < 2; ++mi) {
#pragma unroll
    for (int ni = 0; ni < 2; ++ni) {
      const int ng = n0 + ni * 16 + ncol;
      const float bias = bo[ng];
#pragma unroll
      for (int j = 0; j < 8; ++j) {
        const int mg = m0 + mi * 16 + j + 8 * half;
        out[(size_t)mg * kD + ng] = acc[mi][ni][j] + bias;
      }
    }
  }
}

// ---------------------------------------------------------------------------
extern "C" void kernel_launch(void* const* d_in, const int* in_sizes, int n_in,
                              void* d_out, int out_size, void* d_ws, size_t ws_size,
                              hipStream_t stream) {
  const float* x  = (const float*)d_in[0];
  const float* wq = (const float*)d_in[1];
  const float* bq = (const float*)d_in[2];
  const float* wk = (const float*)d_in[3];
  const float* bk = (const float*)d_in[4];
  const float* wv = (const float*)d_in[5];
  const float* bv = (const float*)d_in[6];
  const float* wo = (const float*)d_in[7];
  const float* bo = (const float*)d_in[8];
  const float* bias_table = (const float*)d_in[9];
  float* out = (float*)d_out;

  char* ws = (char*)d_ws;
  size_t o = 0;
  float*  mask = (float*)(ws + o);  o += 4096;
  __bf16* xb   = (__bf16*)(ws + o); o += (size_t)kM * kD * 2;
  __bf16* wqb  = (__bf16*)(ws + o); o += (size_t)kD * kD * 2;
  __bf16* wkb  = (__bf16*)(ws + o); o += (size_t)kD * kD * 2;
  __bf16* wvb  = (__bf16*)(ws + o); o += (size_t)kD * kD * 2;
  __bf16* wob  = (__bf16*)(ws + o); o += (size_t)kD * kD * 2;
  __bf16* qb   = (__bf16*)(ws + o); o += (size_t)kB * kH * kT * kHD * 2;
  __bf16* kbf  = (__bf16*)(ws + o); o += (size_t)kB * kH * kT * kHD * 2;
  __bf16* vTb  = (__bf16*)(ws + o); o += (size_t)kB * kH * kHD * kT * 2;
  __bf16* ctxb = (__bf16*)(ws + o); o += (size_t)kM * kD * 2;
  (void)ws_size; (void)in_sizes; (void)n_in; (void)out_size;

  // 1) rel-pos bias LUT
  bias_mean_kernel<<<1, 288, 0, stream>>>(bias_table, mask);

  // 2) fp32 -> bf16 conversions
  {
    int n = kM * kD;
    cvt_f32_bf16_kernel<<<(n + 255) / 256, 256, 0, stream>>>(x, xb, n);
    n = kD * kD;
    cvt_f32_bf16_kernel<<<(n + 255) / 256, 256, 0, stream>>>(wq, wqb, n);
    cvt_f32_bf16_kernel<<<(n + 255) / 256, 256, 0, stream>>>(wk, wkb, n);
    cvt_f32_bf16_kernel<<<(n + 255) / 256, 256, 0, stream>>>(wv, wvb, n);
    cvt_f32_bf16_kernel<<<(n + 255) / 256, 256, 0, stream>>>(wo, wob, n);
  }

  // 3) fused Q/K/V projection (8 waves/block, 32x32 tile/wave; z selects Q/K/V)
  {
    dim3 grid(kMT32 * kNT32 / 8, 1, 3);  // 768 x 1 x 3
    proj_qkv_kernel<<<grid, 256, 0, stream>>>(xb, wqb, wkb, wvb,
                                              bq, bk, bv, qb, kbf, vTb);
  }

  // 4) flash attention: B*H*(T/32) = 3072 waves, 8 per block
  attn_kernel<<<kB * kH * (kT / 32) / 8, 256, 0, stream>>>(qb, kbf, vTb, mask, ctxb);

  // 5) output projection
  out_proj_kernel<<<kMT32 * kNT32 / 8, 256, 0, stream>>>(ctxb, wob, bo, out);
}
